// CausalGNNLayer_58007828300539
// MI455X (gfx1250) — matrix operand order repaired and checked by reading, not compiled, not run on hardware
//
#include <hip/hip_runtime.h>

#define IN_CH   512
#define OUT_CH  512
#define NTYPES  4
#define MROWS   64                  // nodes per block (4 x 16-row WMMA tiles)
#define LDS_RS  (IN_CH/2 + 4)       // uint row stride: 256 data + 4 pad (bank rotation)

typedef __attribute__((ext_vector_type(16))) __bf16        v16bf;
typedef __attribute__((ext_vector_type(8)))  float         v8f;
typedef __attribute__((ext_vector_type(4)))  unsigned int  v4u;
typedef __attribute__((ext_vector_type(4)))  float         v4f;

__device__ __forceinline__ unsigned short f2bf(float f) {
    unsigned int u = __builtin_bit_cast(unsigned int, f);
    u += 0x7FFFu + ((u >> 16) & 1u);   // round-to-nearest-even
    return (unsigned short)(u >> 16);
}

// ---------------- partition-by-type kernels ----------------

__global__ void zero_counters(int* ws) {
    if (threadIdx.x < 16) ws[threadIdx.x] = 0;
}

__global__ void type_hist(const int* __restrict__ types, int* __restrict__ counts, int n) {
    int i = blockIdx.x * blockDim.x + threadIdx.x;
    if (i < n) atomicAdd(&counts[types[i] & (NTYPES - 1)], 1);
}

__global__ void scan4(int* ws) {
    // ws[0..3]=counts, ws[4..7]=offsets, ws[8..11]=cursors
    if (threadIdx.x == 0) {
        int acc = 0;
        #pragma unroll
        for (int t = 0; t < NTYPES; ++t) {
            ws[4 + t] = acc;
            ws[8 + t] = acc;
            acc += ws[t];
        }
    }
}

__global__ void scatter_perm(const int* __restrict__ types, int* __restrict__ cursors,
                             int* __restrict__ perm, int n) {
    int i = blockIdx.x * blockDim.x + threadIdx.x;
    if (i < n) {
        int t = types[i] & (NTYPES - 1);
        int pos = atomicAdd(&cursors[t], 1);
        perm[pos] = i;
    }
}

__global__ void convert_w(const float* __restrict__ W, unsigned short* __restrict__ Wb, int n) {
    int i = blockIdx.x * blockDim.x + threadIdx.x;
    if (i < n) Wb[i] = f2bf(W[i]);
}

// -------- grouped GEMM: 64 nodes x 128 cols per block, 4x B reuse --------

__global__ __launch_bounds__(256) void
gemm_tiles(const float* __restrict__ x, const float* __restrict__ bias,
           const unsigned short* __restrict__ Wb,
           const int* __restrict__ counts, const int* __restrict__ offsets,
           const int* __restrict__ perm, float* __restrict__ out) {
    __shared__ unsigned int ldsA[MROWS * LDS_RS];     // 64 rows of 512 bf16 (padded)
    __shared__ __align__(16) int ldsNode[MROWS];

    // map blockIdx.x -> (type, local tile within type)
    int tile = blockIdx.x;
    int type = -1, local = 0, acc = 0;
    #pragma unroll
    for (int t = 0; t < NTYPES; ++t) {
        int nt = (counts[t] + MROWS - 1) / MROWS;
        if (type < 0 && tile < acc + nt) { type = t; local = tile - acc; }
        acc += nt;
    }
    if (type < 0) return;
    const int cnt = counts[type];
    const int off = offsets[type];
    const int vcnt = cnt - local * MROWS;    // valid rows in this tile

    // ---- stage x tile (64 x 512) into LDS as bf16 ----
    const int tid = threadIdx.x;
    #pragma unroll
    for (int pass = 0; pass < 4; ++pass) {
        int row = pass * 16 + (tid >> 4);    // 0..63
        int seg = tid & 15;                  // 16 threads per row, 32 floats each
        bool rvalid = row < vcnt;
        int node = 0;
        if (rvalid) node = perm[off + local * MROWS + row];
        if (seg == 0) ldsNode[row] = node;

        unsigned int* dstp = ldsA + row * LDS_RS + seg * 16;
        if (rvalid) {
            const float* xs = x + (size_t)node * IN_CH + seg * 32;
            #pragma unroll
            for (int i = 0; i < 8; ++i) {
                v4f f = *(const v4f*)(xs + i * 4);
                dstp[i * 2 + 0] = ((unsigned int)f2bf(f.y) << 16) | f2bf(f.x);
                dstp[i * 2 + 1] = ((unsigned int)f2bf(f.w) << 16) | f2bf(f.z);
            }
        } else {
            #pragma unroll
            for (int i = 0; i < 16; ++i) dstp[i] = 0;
        }
    }
    __syncthreads();

    // ---- per-wave: 4 x (16x16) WMMA tiles sharing each B fragment ----
    const int lane = tid & 31;
    const int wave = tid >> 5;
    const int colBase = blockIdx.y * 128 + wave * 16;
    const int col  = lane & 15;
    const int half = lane >> 4;

    const float bv = bias[type * OUT_CH + colBase + col];
    v8f c[4];
    #pragma unroll
    for (int mt = 0; mt < 4; ++mt)
        #pragma unroll
        for (int i = 0; i < 8; ++i) c[mt][i] = bv;

    const unsigned int* wrow =
        (const unsigned int*)(Wb + ((size_t)type * OUT_CH + colBase + col) * IN_CH);
    const unsigned int* arow = ldsA + (lane & 15) * LDS_RS + half * 4;

    #pragma unroll 2
    for (int kb = 0; kb < IN_CH; kb += 32) {
        union { v4u u[2]; v16bf v; } B;
        // B: lane holds col N=lane&15; K = kb+half*8..+7, kb+16+half*8..+7
        const unsigned int* bp = wrow + (kb >> 1) + half * 4;
        B.u[0] = *(const v4u*)(bp);
        B.u[1] = *(const v4u*)(bp + 8);
        #pragma unroll
        for (int mt = 0; mt < 4; ++mt) {
            union { v4u u[2]; v16bf v; } A;
            // A: lane holds row M=lane&15 of subtile mt; K = kb+half*8..+7, kb+16+half*8..+7
            const unsigned int* ap = arow + mt * 16 * LDS_RS + (kb >> 1);
            A.u[0] = *(const v4u*)(ap);
            A.u[1] = *(const v4u*)(ap + 8);
            c[mt] = __builtin_amdgcn_wmma_f32_16x16x32_bf16(false, A.v, false, B.v,
                                                            (short)0, c[mt], false, false);
        }
    }

    // ---- store D tiles (VGPR r -> row half*8+r, lane -> col) ----
    float* obase = out + colBase + col;
    #pragma unroll
    for (int mt = 0; mt < 4; ++mt) {
        union { v4u u[2]; int id[8]; } nid;
        const unsigned int* np = (const unsigned int*)ldsNode + mt * 16 + half * 8;
        nid.u[0] = *(const v4u*)(np);
        nid.u[1] = *(const v4u*)(np + 4);
        #pragma unroll
        for (int r = 0; r < 8; ++r) {
            int m = mt * 16 + half * 8 + r;
            if (m < vcnt)
                obase[(size_t)nid.id[r] * OUT_CH] = c[mt][r];
        }
    }
}

// ---------------- launcher ----------------

extern "C" void kernel_launch(void* const* d_in, const int* in_sizes, int n_in,
                              void* d_out, int out_size, void* d_ws, size_t ws_size,
                              hipStream_t stream) {
    const float* x     = (const float*)d_in[0];
    // d_in[1] = edge_index (unused by the reference forward)
    const int*   types = (const int*)d_in[2];
    const float* W     = (const float*)d_in[3];
    const float* b     = (const float*)d_in[4];
    float*       out   = (float*)d_out;

    const int N = in_sizes[2];              // NUM_NODES
    const int wElems = NTYPES * OUT_CH * IN_CH;

    // workspace layout
    int* counters = (int*)d_ws;             // [0..3]=counts [4..7]=offsets [8..11]=cursors
    int* counts   = counters;
    int* offsets  = counters + 4;
    int* cursors  = counters + 8;
    int* perm     = counters + 16;          // byte offset 64
    size_t wOff   = (64 + (size_t)N * 4 + 255) & ~(size_t)255;
    unsigned short* Wb = (unsigned short*)((char*)d_ws + wOff);

    const int TB = 256;
    zero_counters<<<1, 32, 0, stream>>>(counters);
    type_hist<<<(N + TB - 1) / TB, TB, 0, stream>>>(types, counts, N);
    scan4<<<1, 32, 0, stream>>>(counters);
    scatter_perm<<<(N + TB - 1) / TB, TB, 0, stream>>>(types, cursors, perm, N);
    convert_w<<<(wElems + TB - 1) / TB, TB, 0, stream>>>(W, Wb, wElems);

    dim3 grid((N + MROWS - 1) / MROWS + NTYPES, OUT_CH / 128);
    gemm_tiles<<<grid, TB, 0, stream>>>(x, b, Wb, counts, offsets, perm, out);
}
